// GCN_7035156431540
// MI455X (gfx1250) — compile-verified, matrix operand order checked by reading, not run on hardware
//
#include <hip/hip_runtime.h>
#include <hip/hip_bf16.h>

#define NN     100000
#define EE     1600000
#define FDIM   128
#define MTILES (NN / 16)       // 6250, exact
#define BN_EPS 1e-5f

typedef __bf16 bf16;
typedef bf16  bf16x8  __attribute__((ext_vector_type(8)));
typedef bf16  bf16x16 __attribute__((ext_vector_type(16)));
typedef float v8f     __attribute__((ext_vector_type(8)));

// ---------------------------------------------------------------- utilities
__global__ void k_fill_f32(float* p, int n, float v) {
    int i = blockIdx.x * blockDim.x + threadIdx.x;
    if (i < n) p[i] = v;
}

// degree: deg starts at 1.0 (self loop), add 1 per incoming edge
__global__ void k_deg(const long long* __restrict__ dst, float* __restrict__ deg) {
    int e = blockIdx.x * blockDim.x + threadIdx.x;
    if (e < EE)
        __hip_atomic_fetch_add(&deg[dst[e]], 1.0f, __ATOMIC_RELAXED, __HIP_MEMORY_SCOPE_AGENT);
}

__global__ void k_dinv(float* d) {   // deg >= 1 always (self loops)
    int i = blockIdx.x * blockDim.x + threadIdx.x;
    if (i < NN) d[i] = rsqrtf(d[i]);
}

__global__ void k_cvt_bf16(const float* __restrict__ x, bf16* __restrict__ y, int n) {
    int i = blockIdx.x * blockDim.x + threadIdx.x;
    if (i < n) y[i] = (bf16)x[i];
}

// Repack 128x128 row-major f32 W into per-lane WMMA B-fragment order (bf16).
// Fragment (nt,kk): lane L holds B[k][n] for n = nt*16 + (L&15),
//   h[0] j=0..7 : k = kk*32 +      (L>>4)*8 + j
//   h[1] j=0..7 : k = kk*32 + 16 + (L>>4)*8 + j
__global__ void k_pack_w(const float* __restrict__ W, bf16* __restrict__ Wp) {
    int idx = blockIdx.x * blockDim.x + threadIdx.x;   // 0..16383
    if (idx >= 128 * 128) return;
    int j    = idx & 15;
    int lane = (idx >> 4) & 31;
    int kk   = (idx >> 9) & 3;
    int nt   = idx >> 11;
    int t = j & 7, half = j >> 3;
    int k = kk * 32 + half * 16 + (lane >> 4) * 8 + t;
    int n = nt * 16 + (lane & 15);
    Wp[idx] = (bf16)W[k * 128 + n];
}

// -------------------------------------------------- WMMA GEMM: H = A(bf16) @ Wp
// One wave computes a 16x128 output strip. 4 K-steps x 8 N-tiles = 32 WMMAs.
__global__ void k_gemm_bf16(const bf16* __restrict__ A, const bf16* __restrict__ Wp,
                            float* __restrict__ H) {
    int wave = (blockIdx.x * blockDim.x + threadIdx.x) >> 5;   // M-tile id
    int lane = threadIdx.x & 31;
    if (wave >= MTILES) return;                                 // whole-wave exit

    int m  = lane & 15;
    int kh = lane >> 4;                                         // K half select
    const bf16* arow = A + (size_t)(wave * 16 + m) * FDIM;

    v8f acc[8] = {};
    union Frag { bf16x16 v; bf16x8 h[2]; };

#pragma unroll
    for (int kk = 0; kk < 4; ++kk) {
        Frag a;
        // ISA 16-bit A 16x32 layout: lanes<16 -> K 0..7 / 16..23, lanes>=16 -> +8
        a.h[0] = *(const bf16x8*)(arow + kk * 32 + kh * 8);
        a.h[1] = *(const bf16x8*)(arow + kk * 32 + 16 + kh * 8);
#pragma unroll
        for (int nt = 0; nt < 8; ++nt) {
            Frag b;
            const bf16* bp = Wp + ((size_t)(nt * 4 + kk) * 32 + lane) * 16;
            b.h[0] = *(const bf16x8*)(bp);
            b.h[1] = *(const bf16x8*)(bp + 8);
            acc[nt] = __builtin_amdgcn_wmma_f32_16x16x32_bf16(
                false, a.v, false, b.v, (short)0, acc[nt], false, false);
        }
    }
    // C/D layout: VGPR r -> M=r (lanes 0-15), M=8+r (lanes 16-31); N = lane&15
#pragma unroll
    for (int nt = 0; nt < 8; ++nt) {
#pragma unroll
        for (int r = 0; r < 8; ++r) {
            int mm  = (lane < 16) ? r : (r + 8);
            int col = nt * 16 + (lane & 15);
            H[(size_t)(wave * 16 + mm) * FDIM + col] = acc[nt][r];
        }
    }
}

// ------------------------------------------------------ aggregation (scatter)
// self-loop term initializes the accumulator: Hagg = H * dinv^2
__global__ void k_self_init(const float* __restrict__ H, const float* __restrict__ dinv,
                            float* __restrict__ Hagg) {
    size_t idx = (size_t)blockIdx.x * blockDim.x + threadIdx.x;
    if (idx < (size_t)NN * FDIM) {
        int i = (int)(idx >> 7);
        float d = dinv[i];
        Hagg[idx] = H[idx] * d * d;
    }
}

// wave per edge: gather 512B row from L2-resident H, scale, fp32 HW atomics
__global__ void k_scatter128(const long long* __restrict__ ei, const float* __restrict__ dinv,
                             const float* __restrict__ H, float* __restrict__ Hagg) {
    int wid  = (blockIdx.x * blockDim.x + threadIdx.x) >> 5;
    int lane = threadIdx.x & 31;
    if (wid >= EE) return;
    long long s = ei[wid];
    long long d = ei[EE + wid];
    float c = dinv[s] * dinv[d];
    const float4* hs = (const float4*)(H + (size_t)s * FDIM);
    float4 v = hs[lane];
    float* od = Hagg + (size_t)d * FDIM + lane * 4;
    __hip_atomic_fetch_add(od + 0, v.x * c, __ATOMIC_RELAXED, __HIP_MEMORY_SCOPE_AGENT);
    __hip_atomic_fetch_add(od + 1, v.y * c, __ATOMIC_RELAXED, __HIP_MEMORY_SCOPE_AGENT);
    __hip_atomic_fetch_add(od + 2, v.z * c, __ATOMIC_RELAXED, __HIP_MEMORY_SCOPE_AGENT);
    __hip_atomic_fetch_add(od + 3, v.w * c, __ATOMIC_RELAXED, __HIP_MEMORY_SCOPE_AGENT);
}

// bias + BN(eval) + ReLU, emit bf16 as next layer's A operand
__global__ void k_post(const float* __restrict__ Hagg, const float* __restrict__ b,
                       const float* __restrict__ gamma, const float* __restrict__ beta,
                       const float* __restrict__ rm, const float* __restrict__ rv,
                       bf16* __restrict__ out) {
    size_t idx = (size_t)blockIdx.x * blockDim.x + threadIdx.x;
    if (idx >= (size_t)NN * FDIM) return;
    int f = (int)(idx & (FDIM - 1));
    float v = Hagg[idx] + b[f];
    v = (v - rm[f]) * rsqrtf(rv[f] + BN_EPS) * gamma[f] + beta[f];
    out[idx] = (bf16)fmaxf(v, 0.0f);
}

// ----------------------------------------------------------- layer 3 (Nout=2)
__global__ void k_lin3(const bf16* __restrict__ h, const float* __restrict__ W3,
                       float* __restrict__ H3) {
    int i = blockIdx.x * blockDim.x + threadIdx.x;
    if (i >= NN) return;
    const bf16* r = h + (size_t)i * FDIM;
    float a0 = 0.f, a1 = 0.f;
#pragma unroll 8
    for (int f = 0; f < FDIM; ++f) {
        float x = (float)r[f];
        a0 += x * W3[f * 2 + 0];
        a1 += x * W3[f * 2 + 1];
    }
    H3[i * 2 + 0] = a0;
    H3[i * 2 + 1] = a1;
}

__global__ void k_out_init(const float* __restrict__ H3, const float* __restrict__ dinv,
                           const float* __restrict__ b3, float* __restrict__ out) {
    int i = blockIdx.x * blockDim.x + threadIdx.x;
    if (i >= NN) return;
    float dd = dinv[i] * dinv[i];
    out[i * 2 + 0] = b3[0] + H3[i * 2 + 0] * dd;
    out[i * 2 + 1] = b3[1] + H3[i * 2 + 1] * dd;
}

__global__ void k_scatter2(const long long* __restrict__ ei, const float* __restrict__ dinv,
                           const float* __restrict__ H3, float* __restrict__ out) {
    int e = blockIdx.x * blockDim.x + threadIdx.x;
    if (e >= EE) return;
    long long s = ei[e];
    long long d = ei[EE + e];
    float c = dinv[s] * dinv[d];
    __hip_atomic_fetch_add(&out[d * 2 + 0], H3[s * 2 + 0] * c, __ATOMIC_RELAXED, __HIP_MEMORY_SCOPE_AGENT);
    __hip_atomic_fetch_add(&out[d * 2 + 1], H3[s * 2 + 1] * c, __ATOMIC_RELAXED, __HIP_MEMORY_SCOPE_AGENT);
}

// =========================================================================
extern "C" void kernel_launch(void* const* d_in, const int* in_sizes, int n_in,
                              void* d_out, int out_size, void* d_ws, size_t ws_size,
                              hipStream_t stream) {
    const float*     x  = (const float*)d_in[0];
    const long long* ei = (const long long*)d_in[1];
    const float* W1 = (const float*)d_in[2];
    const float* b1 = (const float*)d_in[3];
    const float* g1 = (const float*)d_in[4];
    const float* be1= (const float*)d_in[5];
    const float* rm1= (const float*)d_in[6];
    const float* rv1= (const float*)d_in[7];
    const float* W2 = (const float*)d_in[8];
    const float* b2 = (const float*)d_in[9];
    const float* g2 = (const float*)d_in[10];
    const float* be2= (const float*)d_in[11];
    const float* rm2= (const float*)d_in[12];
    const float* rv2= (const float*)d_in[13];
    const float* W3 = (const float*)d_in[14];
    const float* b3 = (const float*)d_in[15];
    float* out = (float*)d_out;

    // workspace carve-out (256B aligned)
    char*  ws  = (char*)d_ws;
    size_t off = 0;
    auto carve = [&](size_t bytes) -> void* {
        void* p = ws + off;
        off += (bytes + 255) & ~(size_t)255;
        return p;
    };
    float* dinv = (float*)carve((size_t)NN * 4);
    bf16*  Hbf  = (bf16*) carve((size_t)NN * FDIM * 2);
    float* H    = (float*)carve((size_t)NN * FDIM * 4);
    float* Hagg = (float*)carve((size_t)NN * FDIM * 4);
    bf16*  Wp   = (bf16*) carve((size_t)128 * 128 * 2);
    float* H3   = (float*)carve((size_t)NN * 2 * 4);
    (void)ws_size; (void)in_sizes; (void)n_in; (void)out_size;

    const int T = 256;
    const size_t NF = (size_t)NN * FDIM;

    // ---- norm = dinv[src]*dinv[dst], deg includes self loop
    k_fill_f32<<<(NN + T - 1) / T, T, 0, stream>>>(dinv, NN, 1.0f);
    k_deg     <<<(EE + T - 1) / T, T, 0, stream>>>(ei + EE, dinv);
    k_dinv    <<<(NN + T - 1) / T, T, 0, stream>>>(dinv);

    // ---- A operand for layer 1
    k_cvt_bf16<<<(int)((NF + T - 1) / T), T, 0, stream>>>(x, Hbf, (int)NF);

    const int gemmBlocks = (MTILES + 3) / 4;          // 4 waves / 128-thread block
    const int scatBlocks = (EE + 7) / 8;              // 8 waves / 256-thread block

    // ================= layer 1 =================
    k_pack_w   <<<(128 * 128) / T, T, 0, stream>>>(W1, Wp);
    k_gemm_bf16<<<gemmBlocks, 128, 0, stream>>>(Hbf, Wp, H);
    k_self_init<<<(int)((NF + T - 1) / T), T, 0, stream>>>(H, dinv, Hagg);
    k_scatter128<<<scatBlocks, T, 0, stream>>>(ei, dinv, H, Hagg);
    k_post     <<<(int)((NF + T - 1) / T), T, 0, stream>>>(Hagg, b1, g1, be1, rm1, rv1, Hbf);

    // ================= layer 2 =================
    k_pack_w   <<<(128 * 128) / T, T, 0, stream>>>(W2, Wp);
    k_gemm_bf16<<<gemmBlocks, 128, 0, stream>>>(Hbf, Wp, H);
    k_self_init<<<(int)((NF + T - 1) / T), T, 0, stream>>>(H, dinv, Hagg);
    k_scatter128<<<scatBlocks, T, 0, stream>>>(ei, dinv, H, Hagg);
    k_post     <<<(int)((NF + T - 1) / T), T, 0, stream>>>(Hagg, b2, g2, be2, rm2, rv2, Hbf);

    // ================= layer 3 =================
    k_lin3    <<<(NN + T - 1) / T, T, 0, stream>>>(Hbf, W3, H3);
    k_out_init<<<(NN + T - 1) / T, T, 0, stream>>>(H3, dinv, b3, out);
    k_scatter2<<<(EE + T - 1) / T, T, 0, stream>>>(ei, dinv, H3, out);
}
